// VariableGT_42434276884502
// MI455X (gfx1250) — matrix operand order repaired and checked by reading, not compiled, not run on hardware
//
#include <hip/hip_runtime.h>
#include <cstdint>

#define S_LEN 128
#define B_SZ  32
#define V_SZ  36
#define HID   128
#define ST    9
#define NH    8
#define C1    512      // 4*HID
#define CV    4608     // HID*V

typedef __bf16 bf16;
typedef __attribute__((ext_vector_type(8)))  bf16  bf16x8;
typedef __attribute__((ext_vector_type(16))) bf16  bf16x16;
typedef __attribute__((ext_vector_type(8)))  float f32x8;
typedef __attribute__((ext_vector_type(2)))  float f32x2;
typedef int v4i __attribute__((vector_size(16)));

#if __has_builtin(__builtin_amdgcn_global_load_async_to_lds_b128)
#define HAVE_ASYNC_LDS 1
#else
#define HAVE_ASYNC_LDS 0
#endif

#if HAVE_ASYNC_LDS
#if __has_builtin(__builtin_amdgcn_s_wait_asynccnt)
#define WAIT_ASYNC(n) __builtin_amdgcn_s_wait_asynccnt(n)
#else
#define WAIT_ASYNC(n) asm volatile("s_wait_asynccnt %0" ::"i"(n))
#endif
#else
#define WAIT_ASYNC(n) do {} while (0)
#endif

// async 16-byte global -> LDS copy (per lane); fallback = synchronous copy via VGPRs
__device__ __forceinline__ void async_cp16(const bf16* g, bf16* l) {
#if HAVE_ASYNC_LDS
    __builtin_amdgcn_global_load_async_to_lds_b128(
        (__attribute__((address_space(1))) v4i*)g,
        (__attribute__((address_space(3))) v4i*)l,
        0, 0);
#else
    *(bf16x8*)l = *(const bf16x8*)g;
#endif
}

__device__ __forceinline__ float gelu_exact(float x) {
    return 0.5f * x * (1.0f + erff(x * 0.70710678118654752f));
}

// ---------------- positional encoding: PE[b][s][h] ----------------
__global__ void k_pe(const float* __restrict__ Xtimes, float* __restrict__ PE) {
    int i = blockIdx.x * 256 + threadIdx.x;
    if (i >= B_SZ * S_LEN * HID) return;
    int h = i & (HID - 1);
    int s = (i >> 7) & (S_LEN - 1);
    int b = i >> 14;
    float t = Xtimes[s * B_SZ + b];
    float dv = expf(-(float)(h >> 1) * (9.210340371976184f / 64.0f)); // ln(1e4)/64
    float ang = t * dv;
    PE[i] = (h & 1) ? cosf(ang) : sinf(ang);
}

// ---------------- generic f32 -> bf16 ----------------
__global__ void k_f2bf(const float* __restrict__ in, bf16* __restrict__ out, int n) {
    int i = blockIdx.x * 256 + threadIdx.x;
    if (i < n) out[i] = (bf16)in[i];
}

// ---- build transposed, scale-folded QKV weight: Wqkvb[n'][k], n'=which*128+n ----
__global__ void k_prep_qkv(const float* __restrict__ Wq, const float* __restrict__ Wk,
                           const float* __restrict__ Wv, bf16* __restrict__ Wqkvb) {
    int i = blockIdx.x * 256 + threadIdx.x;   // 384*128
    if (i >= 384 * HID) return;
    int k = i & 127;
    int n = i >> 7;                  // 0..383
    int which = n >> 7;
    int nn = n & 127;
    const float* W = (which == 0) ? Wq : (which == 1) ? Wk : Wv;
    float scl = (which < 2) ? 0.25f : 1.0f;   // fold 1/sqrt(DK) into q and k
    Wqkvb[n * HID + k] = (bf16)(W[k * HID + nn] * scl);
}

// ---------------- masked stats + emb + h4 (bf16), one block per (b,v) ----------------
__global__ void k_stats_h4(const float* __restrict__ Xtemp, const float* __restrict__ mask,
                           const float* __restrict__ Xstatic,
                           const float* __restrict__ W_te, const float* __restrict__ b_te,
                           const float* __restrict__ W_emb, const float* __restrict__ b_emb,
                           const float* __restrict__ PE,
                           float* __restrict__ embw, bf16* __restrict__ h4b) {
    int bv = blockIdx.x;                 // B*V = 1152
    int b = bv / V_SZ, v = bv % V_SZ;
    int tid = threadIdx.x;               // 128
    __shared__ float sm[S_LEN];
    __shared__ float r0[128], r1[128];
    float x = Xtemp[(tid * B_SZ + b) * V_SZ + v];
    float m = mask[(tid * B_SZ + b) * V_SZ + v];
    sm[tid] = m;
    r0[tid] = x * m;
    r1[tid] = m;
    __syncthreads();
    for (int off = 64; off > 0; off >>= 1) {
        if (tid < off) { r0[tid] += r0[tid + off]; r1[tid] += r1[tid + off]; }
        __syncthreads();
    }
    float sumxm = r0[0], cnt = r1[0];
    // tid == h from here
    const float* peb = PE + (size_t)b * S_LEN * HID + tid;
    float psum = 0.f;
    for (int s = 0; s < S_LEN; s++) psum += peb[s * HID] * sm[s];
    float e = b_emb[v * HID + tid];
    const float* we = W_emb + (size_t)(v * HID + tid) * ST;
    const float* xs = Xstatic + b * ST;
    #pragma unroll
    for (int t = 0; t < ST; t++) e += xs[t] * we[t];
    embw[bv * HID + tid] = e;
    float h2 = W_te[v * HID + tid] * sumxm + b_te[v * HID + tid] * cnt + psum + e * cnt;
    h4b[bv * HID + tid] = (bf16)(h2 / (cnt + 1.0f));
}

// ---------------- QKV GEMM (bf16 WMMA): qkv[m][384] = h4[m][128] @ Wqkv ----------------
__global__ void k_qkv_gemm(const bf16* __restrict__ h4b, const bf16* __restrict__ Wqkvb,
                           float* __restrict__ qkv) {
    int tile = blockIdx.x;               // 72*24 tiles, 1 wave each
    int mt = tile / 24, nt = tile % 24;
    int l = threadIdx.x;
    int lo = l & 15, hi = (l >> 4) & 1;
    int m = mt * 16 + lo;
    int n = nt * 16 + lo;
    f32x8 acc = {};
    for (int kk = 0; kk < 4; kk++) {
        const bf16* pa = h4b + m * HID + kk * 32 + 16 * hi;
        bf16x8 a0 = *(const bf16x8*)pa;
        bf16x8 a1 = *(const bf16x8*)(pa + 8);
        bf16x16 a;
        #pragma unroll
        for (int j = 0; j < 8; j++) { a[j] = a0[j]; a[j + 8] = a1[j]; }
        bf16x16 bfr = *(const bf16x16*)(Wqkvb + n * HID + kk * 32 + 16 * hi);
        acc = __builtin_amdgcn_wmma_f32_16x16x32_bf16(false, a, false, bfr, (short)0, acc, false, false);
    }
    #pragma unroll
    for (int r = 0; r < 8; r++)
        qkv[(mt * 16 + r + 8 * hi) * 384 + nt * 16 + lo] = acc[r];
}

// ---------------- attention scores: A[b][h][v][u] ----------------
__global__ void k_scores(const float* __restrict__ qkv, float* __restrict__ Asc) {
    int bh = blockIdx.x;                 // B*NH = 256
    int b = bh >> 3, h = bh & 7;
    for (int p = threadIdx.x; p < V_SZ * V_SZ; p += 256) {
        int v = p / V_SZ, u = p % V_SZ;
        const float* qp = qkv + (b * V_SZ + v) * 384 + h * 16;
        const float* kp = qkv + (b * V_SZ + u) * 384 + 128 + h * 16;
        float s = 0.f;
        #pragma unroll
        for (int d = 0; d < 16; d++) s += qp[d] * kp[d];
        Asc[((bh) * V_SZ + v) * V_SZ + u] = s;
    }
}

// ------- fused softmax + attention-apply (f32 WMMA 16x16x4) + meanv, -> v5t bf16 -------
// one block (256 thr = 8 waves, wave w = head w) per (b,s); v5t layout [b][s][ch]
__global__ void k_attn_apply(const float* __restrict__ Xtemp, const float* __restrict__ mask,
                             const float* __restrict__ W_te, const float* __restrict__ b_te,
                             const float* __restrict__ embw, const float* __restrict__ PE,
                             const float* __restrict__ Asc, const float* __restrict__ qkv,
                             bf16* __restrict__ v5t) {
    int bs = blockIdx.x;
    int b = bs >> 7, s = bs & 127;
    int tid = threadIdx.x;
    __shared__ float Hcol[HID * V_SZ];              // [c][u]  (18 KB)
    __shared__ float Pt[NH * V_SZ * V_SZ + 48];     // [h][u][v] + overrun pad (~41.7 KB)
    __shared__ float xt[V_SZ], mk[V_SZ], pe_s[HID];
    if (tid < V_SZ) {
        xt[tid] = Xtemp[(s * B_SZ + b) * V_SZ + tid];
        mk[tid] = mask[(s * B_SZ + b) * V_SZ + tid];
    }
    if (tid < HID) pe_s[tid] = PE[((size_t)b * S_LEN + s) * HID + tid];
    __syncthreads();
    // H column for this (b,s): H[c][u]
    for (int i = tid; i < HID * V_SZ; i += 256) {
        int c = i & 127, u = i >> 7;
        Hcol[c * V_SZ + u] = xt[u] * W_te[u * HID + c] + b_te[u * HID + c]
                           + pe_s[c] + embw[(b * V_SZ + u) * HID + c];
    }
    // softmax rows (h,v) over u, write P^T[h][u][v]
    for (int r = tid; r < NH * V_SZ; r += 256) {
        int h = r / V_SZ, v = r % V_SZ;
        const float* arow = Asc + (((b * 8 + h) * V_SZ) + v) * V_SZ;
        float mx = -1e30f;
        for (int u = 0; u < V_SZ; u++) {
            float lv = arow[u] + (mk[u] > 0.5f ? 0.f : -1e6f);
            mx = fmaxf(mx, lv);
        }
        float sum = 0.f;
        for (int u = 0; u < V_SZ; u++) {
            float e = expf(arow[u] + (mk[u] > 0.5f ? 0.f : -1e6f) - mx);
            Pt[(h * V_SZ + u) * V_SZ + v] = e;
            sum += e;
        }
        float inv = 1.f / sum;
        for (int u = 0; u < V_SZ; u++) Pt[(h * V_SZ + u) * V_SZ + v] *= inv;
    }
    __syncthreads();
    // per head: D(16x48) = Hcol_head(16x36) @ P^T(36x48), K=36 = 9 steps of x4
    int w  = tid >> 5;                    // head
    int l  = tid & 31;
    int lo = l & 15, hi = (l >> 4) & 1;
    f32x8 acc[3] = {};
    const float* hc = Hcol + (w * 16 + lo) * V_SZ;
    for (int t = 0; t < 9; t++) {
        int u0 = 4 * t + 2 * hi;
        f32x2 a;
        a.x = hc[u0];
        a.y = hc[u0 + 1];
        #pragma unroll
        for (int vt = 0; vt < 3; vt++) {
            f32x2 bb;
            bb.x = Pt[(w * V_SZ + u0) * V_SZ + vt * 16 + lo];
            bb.y = Pt[(w * V_SZ + u0 + 1) * V_SZ + vt * 16 + lo];
            acc[vt] = __builtin_amdgcn_wmma_f32_16x16x4_f32(false, a, false, bb, (short)0, acc[vt], false, false);
        }
    }
    // + mean_v, store bf16 to v5t[b][s][ch], ch = (w*16+d)*36 + v
    bf16* outp = v5t + (size_t)(b * S_LEN + s) * CV;
    for (int vt = 0; vt < 3; vt++) {
        int v = vt * 16 + lo;
        if (v >= V_SZ) continue;          // after all WMMA: safe divergence
        const float* mv = qkv + (b * V_SZ + v) * 384 + 256;
        #pragma unroll
        for (int r = 0; r < 8; r++) {
            int c = w * 16 + r + 8 * hi;
            outp[c * V_SZ + v] = (bf16)(acc[vt][r] + mv[c]);
        }
    }
}

// ---------------- main GEMM: y[b](512x128) = W1(512x4608) @ v5[b](4608x128) ----------------
// B panel (shared by all 8 waves) staged through LDS with async copies, double buffered.
#define KCH    256                  // K elements per chunk
#define NCHUNK (CV / KCH)           // 18
#define LROW   264                  // padded LDS row stride (528 B = 132 dw, 132 % 64 = 4)

__global__ void k_gemm1(const bf16* __restrict__ W1b, const bf16* __restrict__ v5t,
                        float* __restrict__ y) {
    __shared__ bf16 Bs[2][16 * LROW];   // 2 x 8.25 KB
    int blk = blockIdx.x;                // 32 b * 8 ntiles
    int b = blk >> 3, nt = blk & 7;
    int tid = threadIdx.x;
    int w = tid >> 5, l = tid & 31;
    int lo = l & 15, hi = (l >> 4) & 1;
    const bf16* gB = v5t + (size_t)(b * S_LEN + nt * 16) * CV;

    auto issue = [&](int c) {            // stage chunk c into Bs[c&1]
        #pragma unroll
        for (int it = 0; it < 2; it++) {
            int idx = tid + it * 256;    // 0..511 : 16 rows x 32 16B-pieces
            int r   = idx >> 5;
            int c16 = idx & 31;
            async_cp16(gB + (size_t)r * CV + c * KCH + c16 * 8,
                       &Bs[c & 1][r * LROW + c16 * 8]);
        }
    };

    f32x8 acc[4] = {};
    issue(0);
    for (int c = 0; c < NCHUNK; c++) {
        if (c + 1 < NCHUNK) { issue(c + 1); WAIT_ASYNC(2); }
        else                { WAIT_ASYNC(0); }
        __syncthreads();
        const bf16* bsrc = &Bs[c & 1][lo * LROW + 16 * hi];
        for (int kkL = 0; kkL < KCH / 32; kkL++) {
            bf16x8 b0 = *(const bf16x8*)(bsrc + kkL * 32);
            bf16x8 b1 = *(const bf16x8*)(bsrc + kkL * 32 + 8);
            bf16x16 bfr;
            #pragma unroll
            for (int j = 0; j < 8; j++) { bfr[j] = b0[j]; bfr[j + 8] = b1[j]; }
            int kk = c * (KCH / 32) + kkL;
            #pragma unroll
            for (int q = 0; q < 4; q++) {
                int m = (w + 8 * q) * 16 + lo;
                const bf16* pa = W1b + (size_t)m * CV + kk * 32 + 16 * hi;
                bf16x8 a0 = *(const bf16x8*)pa;
                bf16x8 a1 = *(const bf16x8*)(pa + 8);
                bf16x16 a;
                #pragma unroll
                for (int j = 0; j < 8; j++) { a[j] = a0[j]; a[j + 8] = a1[j]; }
                acc[q] = __builtin_amdgcn_wmma_f32_16x16x32_bf16(false, a, false, bfr, (short)0, acc[q], false, false);
            }
        }
        __syncthreads();
    }
    #pragma unroll
    for (int q = 0; q < 4; q++)
        for (int r = 0; r < 8; r++) {
            int o = (w + 8 * q) * 16 + r + 8 * hi;
            y[((size_t)b * C1 + o) * S_LEN + nt * 16 + lo] = acc[q][r];
        }
}

// ---------------- BN stats per channel over (B,S); stats[o] = {scale, shift} ----------------
__global__ void k_bnstats(const float* __restrict__ y, const float* __restrict__ g,
                          const float* __restrict__ be, float* __restrict__ stats, int O) {
    int o = blockIdx.x;
    float s1 = 0.f, s2 = 0.f;
    for (int i = threadIdx.x; i < B_SZ * S_LEN; i += 256) {
        int b = i >> 7, s = i & 127;
        float v = y[((size_t)b * O + o) * S_LEN + s];
        s1 += v; s2 += v * v;
    }
    __shared__ float r1[256], r2[256];
    r1[threadIdx.x] = s1; r2[threadIdx.x] = s2;
    __syncthreads();
    for (int off = 128; off > 0; off >>= 1) {
        if (threadIdx.x < off) { r1[threadIdx.x] += r1[threadIdx.x + off]; r2[threadIdx.x] += r2[threadIdx.x + off]; }
        __syncthreads();
    }
    if (threadIdx.x == 0) {
        float mean = r1[0] * (1.0f / 4096.0f);
        float var  = r2[0] * (1.0f / 4096.0f) - mean * mean;
        float sc = rsqrtf(var + 1e-5f) * g[o];
        stats[o * 2]     = sc;
        stats[o * 2 + 1] = be[o] - mean * sc;
    }
}

// -------- normalize + gelu, transpose (B,O,S)->(B,S,O), store bf16 for next GEMM --------
__global__ void k_norm_gelu_bf(const float* __restrict__ y, const float* __restrict__ stats,
                               bf16* __restrict__ yb) {
    int i = blockIdx.x * 256 + threadIdx.x;   // (b*S+s)*C1 + o
    if (i >= B_SZ * S_LEN * C1) return;
    int o = i & (C1 - 1);
    int s = (i >> 9) & 127;
    int b = i >> 16;
    float x = y[((size_t)b * C1 + o) * S_LEN + s];
    x = x * stats[o * 2] + stats[o * 2 + 1];
    yb[i] = (bf16)gelu_exact(x);
}

// ---------------- second GEMM: z[b](128x128) = W2(128x512) @ y'[b](512x128) ----------------
__global__ void k_gemm2(const bf16* __restrict__ W2b, const bf16* __restrict__ yb,
                        float* __restrict__ z) {
    int blk = blockIdx.x;                // 32 b * 8 ntiles
    int b = blk >> 3, nt = blk & 7;
    int w = threadIdx.x >> 5, l = threadIdx.x & 31;
    int lo = l & 15, hi = (l >> 4) & 1;
    const bf16* Bbase = yb + (size_t)(b * S_LEN + nt * 16 + lo) * C1 + 16 * hi;
    int m = w * 16 + lo;
    f32x8 acc = {};
    for (int kk = 0; kk < C1 / 32; kk++) {
        bf16x16 bfr = *(const bf16x16*)(Bbase + kk * 32);
        const bf16* pa = W2b + (size_t)m * C1 + kk * 32 + 16 * hi;
        bf16x8 a0 = *(const bf16x8*)pa;
        bf16x8 a1 = *(const bf16x8*)(pa + 8);
        bf16x16 a;
        #pragma unroll
        for (int j = 0; j < 8; j++) { a[j] = a0[j]; a[j + 8] = a1[j]; }
        acc = __builtin_amdgcn_wmma_f32_16x16x32_bf16(false, a, false, bfr, (short)0, acc, false, false);
    }
    #pragma unroll
    for (int r = 0; r < 8; r++)
        z[((size_t)b * HID + w * 16 + r + 8 * hi) * S_LEN + nt * 16 + lo] = acc[r];
}

// ---------------- final normalize + gelu to output (B,HID,S) fp32 ----------------
__global__ void k_norm_gelu_out(const float* __restrict__ z, const float* __restrict__ stats,
                                float* __restrict__ out) {
    int i = blockIdx.x * 256 + threadIdx.x;
    if (i >= B_SZ * HID * S_LEN) return;
    int o = (i >> 7) & 127;
    float x = z[i] * stats[o * 2] + stats[o * 2 + 1];
    out[i] = gelu_exact(x);
}

// =====================================================================================
extern "C" void kernel_launch(void* const* d_in, const int* in_sizes, int n_in,
                              void* d_out, int out_size, void* d_ws, size_t ws_size,
                              hipStream_t stream) {
    const float* Xtemp   = (const float*)d_in[0];
    const float* Xtimes  = (const float*)d_in[1];
    const float* Xstatic = (const float*)d_in[2];
    const float* mask    = (const float*)d_in[3];
    const float* W_te    = (const float*)d_in[4];
    const float* b_te    = (const float*)d_in[5];
    const float* W_emb   = (const float*)d_in[6];
    const float* b_emb   = (const float*)d_in[7];
    const float* Wq      = (const float*)d_in[8];
    const float* Wk      = (const float*)d_in[9];
    const float* Wv      = (const float*)d_in[10];
    const float* W1      = (const float*)d_in[11];
    // d_in[12] = b1 (cancels under BatchNorm)
    const float* g1      = (const float*)d_in[13];
    const float* be1     = (const float*)d_in[14];
    const float* W2      = (const float*)d_in[15];
    // d_in[16] = b2 (cancels under BatchNorm)
    const float* g2      = (const float*)d_in[17];
    const float* be2     = (const float*)d_in[18];

    uintptr_t p = (uintptr_t)d_ws;
    auto carve = [&](size_t bytes) -> void* {
        void* r = (void*)p;
        p += (bytes + 255) & ~(size_t)255;
        return r;
    };
    float* PE     = (float*)carve((size_t)B_SZ * S_LEN * HID * 4);      // 2 MB
    float* embw   = (float*)carve((size_t)B_SZ * V_SZ * HID * 4);
    bf16*  h4b    = (bf16*) carve((size_t)B_SZ * V_SZ * HID * 2);
    bf16*  Wqkvb  = (bf16*) carve((size_t)384 * HID * 2);
    bf16*  W1b    = (bf16*) carve((size_t)C1 * CV * 2);                 // 4.7 MB
    bf16*  W2b    = (bf16*) carve((size_t)HID * C1 * 2);
    float* qkv    = (float*)carve((size_t)B_SZ * V_SZ * 384 * 4);
    float* Asc    = (float*)carve((size_t)B_SZ * NH * V_SZ * V_SZ * 4);
    bf16*  v5t    = (bf16*) carve((size_t)B_SZ * S_LEN * CV * 2);       // 37.7 MB
    float* ybuf   = (float*)carve((size_t)B_SZ * C1 * S_LEN * 4);       // 8.4 MB
    bf16*  yb     = (bf16*) carve((size_t)B_SZ * S_LEN * C1 * 2);
    float* zbuf   = (float*)carve((size_t)B_SZ * HID * S_LEN * 4);
    float* stats1 = (float*)carve((size_t)C1 * 2 * 4);
    float* stats2 = (float*)carve((size_t)HID * 2 * 4);

    k_pe<<<(B_SZ * S_LEN * HID + 255) / 256, 256, 0, stream>>>(Xtimes, PE);
    k_f2bf<<<(C1 * CV + 255) / 256, 256, 0, stream>>>(W1, W1b, C1 * CV);
    k_f2bf<<<(HID * C1 + 255) / 256, 256, 0, stream>>>(W2, W2b, HID * C1);
    k_prep_qkv<<<(384 * HID + 255) / 256, 256, 0, stream>>>(Wq, Wk, Wv, Wqkvb);
    k_stats_h4<<<B_SZ * V_SZ, 128, 0, stream>>>(Xtemp, mask, Xstatic, W_te, b_te,
                                                W_emb, b_emb, PE, embw, h4b);
    k_qkv_gemm<<<72 * 24, 32, 0, stream>>>(h4b, Wqkvb, qkv);
    k_scores<<<B_SZ * NH, 256, 0, stream>>>(qkv, Asc);
    k_attn_apply<<<B_SZ * S_LEN, 256, 0, stream>>>(Xtemp, mask, W_te, b_te,
                                                   embw, PE, Asc, qkv, v5t);
    k_gemm1<<<B_SZ * 8, 256, 0, stream>>>(W1b, v5t, ybuf);
    k_bnstats<<<C1, 256, 0, stream>>>(ybuf, g1, be1, stats1, C1);
    k_norm_gelu_bf<<<(B_SZ * S_LEN * C1 + 255) / 256, 256, 0, stream>>>(ybuf, stats1, yb);
    k_gemm2<<<B_SZ * 8, 256, 0, stream>>>(W2b, yb, zbuf);
    k_bnstats<<<HID, 256, 0, stream>>>(zbuf, g2, be2, stats2, HID);
    k_norm_gelu_out<<<(B_SZ * HID * S_LEN + 255) / 256, 256, 0, stream>>>(zbuf, stats2, (float*)d_out);
    (void)in_sizes; (void)n_in; (void)out_size; (void)ws_size;
}
